// SelfAttention_62654982914373
// MI455X (gfx1250) — matrix operand order repaired and checked by reading, not compiled, not run on hardware
//
#include <hip/hip_runtime.h>
#include <hip/hip_bf16.h>
#include <math.h>

// Problem constants (from reference)
#define Bb  4
#define Ss  2048
#define Dd  1024
#define Hh  16
#define DHh 64
#define WPR (Ss / 32)   // mask words per row

typedef __attribute__((ext_vector_type(16))) _Float16 v16h;
typedef __attribute__((ext_vector_type(8)))  _Float16 v8h;
typedef __attribute__((ext_vector_type(8)))  float    v8f;

// ---------------------------------------------------------------------------
// WMMA fragment helpers (16x16x32 f16, wave32; A/B k order below is this file's own, not AMD's table order)
//
// A (16x32 f16): lane L holds row M=L%16, elements i = K=(L>=16?16:0)+i:
// one contiguous 16-half run at offset off = (L>=16)?16:0.
// B (32x16 f16): lane L holds col N=L%16, elements i = K=(L>=16?16:0)+i:
// one contiguous 16-half run.
// C/D (16x16 f32): lane L holds col N=L%16; VGPR g holds row M=g+(L>=16?8:0).
// ---------------------------------------------------------------------------
static __device__ inline v16h load_a_frag(const _Float16* rowbase, int lane) {
  const int off = (lane & 16) ? 16 : 0;
  v8h lo = *(const v8h*)(rowbase + off);
  v8h hi = *(const v8h*)(rowbase + off + 8);
  v16h r;
#pragma unroll
  for (int i = 0; i < 8; ++i) { r[i] = lo[i]; r[i + 8] = hi[i]; }
  return r;
}

static __device__ inline v16h load_b_frag(const _Float16* p) {
  v8h lo = *(const v8h*)(p);
  v8h hi = *(const v8h*)(p + 8);
  v16h r;
#pragma unroll
  for (int i = 0; i < 8; ++i) { r[i] = lo[i]; r[i + 8] = hi[i]; }
  return r;
}

static __device__ inline v8f wmma_f16(v16h a, v16h b, v8f c) {
  return __builtin_amdgcn_wmma_f32_16x16x32_f16(false, a, false, b,
                                                (short)0, c, false, false);
}

// Async global -> LDS copy, 16 bytes per lane (ASYNCcnt-tracked).
static __device__ inline void async_copy_b128(unsigned lds_off, const void* gptr) {
  asm volatile("global_load_async_to_lds_b128 %0, %1, off"
               :: "v"(lds_off), "v"(gptr) : "memory");
}

// ---------------------------------------------------------------------------
// fp32 -> f16 weight conversion (Wq|Wk|Wv|Wo packed contiguously in ws)
// ---------------------------------------------------------------------------
__global__ __launch_bounds__(256)
void cvt_w_kernel(const float* __restrict__ wq, const float* __restrict__ wk,
                  const float* __restrict__ wv, const float* __restrict__ wo,
                  _Float16* __restrict__ dst) {
  const int n = Dd * Dd;                     // 1M, power of two
  int i = blockIdx.x * blockDim.x + threadIdx.x;
  const float* src = (i < n) ? wq : (i < 2 * n) ? wk : (i < 3 * n) ? wv : wo;
  dst[i] = (_Float16)src[i & (n - 1)];
}

// ---------------------------------------------------------------------------
// mask[b,q,k] (int32) -> packed bits (bit j of word = mask[.., kw*32+j] != 0)
// ---------------------------------------------------------------------------
__global__ __launch_bounds__(256)
void pack_mask_kernel(const int* __restrict__ mask, unsigned* __restrict__ out) {
  int w = blockIdx.x * blockDim.x + threadIdx.x;   // Bb*Ss*WPR words
  int bq = w / WPR, kw = w % WPR;
  const int* p = mask + (size_t)bq * Ss + kw * 32;
  unsigned v = 0;
#pragma unroll
  for (int j = 0; j < 32; ++j) v |= (p[j] != 0 ? 1u : 0u) << j;
  out[w] = v;
}

// ---------------------------------------------------------------------------
// 64x64-tile GEMM: D = alpha * (A @ B + bias). A fp32 (A_F16=0) or f16.
// B is f16 [K,N] row-major. Store modes:
//   0: f16 to [B,H,S,DH]      (Q/K head-split layout)
//   1: f16 to [B,H,DH,S]      (V transposed for PV B-fragments)
//   2: f32 to [M,N] row-major (final output)
// Block = 128 threads (4 waves); each wave owns a 32x32 quadrant (2x2 WMMA).
// ---------------------------------------------------------------------------
template <int A_F16>
__global__ __launch_bounds__(128)
void gemm64_kernel(const void* __restrict__ Aptr, const _Float16* __restrict__ Bh,
                   const float* __restrict__ bias, void* __restrict__ dst,
                   int M, int N, int K, int mode, float alpha) {
  __shared__ _Float16 Ash[64 * 40];    // [row][k], padded stride 40 halfs
  __shared__ _Float16 BTsh[64 * 40];   // transposed: [n][k], padded

  const int t = threadIdx.x;
  const int mTiles = M >> 6;
  const int m0 = (blockIdx.x % mTiles) << 6;
  const int n0 = (blockIdx.x / mTiles) << 6;
  const int wave = t >> 5, lane = t & 31;
  const int rw = (wave >> 1) & 1, cw = wave & 1;

  v8f acc[2][2];
#pragma unroll
  for (int mi = 0; mi < 2; ++mi)
#pragma unroll
    for (int ni = 0; ni < 2; ++ni)
#pragma unroll
      for (int g = 0; g < 8; ++g) acc[mi][ni][g] = 0.0f;

  const int lrowA = t >> 1, lhalf = t & 1;   // A loader: row 0..63, 16-col half
  const int lkB = t >> 2, lngB = t & 3;      // B loader: k-row 0..31, 16-col grp

  for (int k0 = 0; k0 < K; k0 += 32) {
    // ---- stage A tile (64x32) into LDS as f16 ----
    if (A_F16) {
      const _Float16* Ar =
          (const _Float16*)Aptr + (size_t)(m0 + lrowA) * K + k0 + lhalf * 16;
      *(v8h*)&Ash[lrowA * 40 + lhalf * 16]     = *(const v8h*)Ar;
      *(v8h*)&Ash[lrowA * 40 + lhalf * 16 + 8] = *(const v8h*)(Ar + 8);
    } else {
      const float* Ar =
          (const float*)Aptr + (size_t)(m0 + lrowA) * K + k0 + lhalf * 16;
      float4 f0 = ((const float4*)Ar)[0];
      float4 f1 = ((const float4*)Ar)[1];
      float4 f2 = ((const float4*)Ar)[2];
      float4 f3 = ((const float4*)Ar)[3];
      v8h h0, h1;
      h0[0] = (_Float16)f0.x; h0[1] = (_Float16)f0.y;
      h0[2] = (_Float16)f0.z; h0[3] = (_Float16)f0.w;
      h0[4] = (_Float16)f1.x; h0[5] = (_Float16)f1.y;
      h0[6] = (_Float16)f1.z; h0[7] = (_Float16)f1.w;
      h1[0] = (_Float16)f2.x; h1[1] = (_Float16)f2.y;
      h1[2] = (_Float16)f2.z; h1[3] = (_Float16)f2.w;
      h1[4] = (_Float16)f3.x; h1[5] = (_Float16)f3.y;
      h1[6] = (_Float16)f3.z; h1[7] = (_Float16)f3.w;
      *(v8h*)&Ash[lrowA * 40 + lhalf * 16]     = h0;
      *(v8h*)&Ash[lrowA * 40 + lhalf * 16 + 8] = h1;
    }
    // ---- stage B tile (32x64) transposed into LDS ----
    {
      const _Float16* Br = Bh + (size_t)(k0 + lkB) * N + n0 + lngB * 16;
      v8h b0 = *(const v8h*)Br;
      v8h b1 = *(const v8h*)(Br + 8);
#pragma unroll
      for (int j = 0; j < 8; ++j) {
        BTsh[(lngB * 16 + j) * 40 + lkB]     = b0[j];
        BTsh[(lngB * 16 + 8 + j) * 40 + lkB] = b1[j];
      }
      if (k0 + 32 < K)  // hint next tiles into cache (global_prefetch_b8)
        __builtin_prefetch(Bh + (size_t)(k0 + 32 + lkB) * N + n0 + lngB * 16, 0, 1);
    }
    __syncthreads();

    v16h afrag[2], bfrag[2];
#pragma unroll
    for (int mi = 0; mi < 2; ++mi)
      afrag[mi] = load_a_frag(&Ash[(rw * 32 + mi * 16 + (lane & 15)) * 40], lane);
#pragma unroll
    for (int ni = 0; ni < 2; ++ni)
      bfrag[ni] = load_b_frag(
          &BTsh[(cw * 32 + ni * 16 + (lane & 15)) * 40 + ((lane & 16) ? 16 : 0)]);
#pragma unroll
    for (int mi = 0; mi < 2; ++mi)
#pragma unroll
      for (int ni = 0; ni < 2; ++ni)
        acc[mi][ni] = wmma_f16(afrag[mi], bfrag[ni], acc[mi][ni]);
    __syncthreads();
  }

  // ---- epilogue ----
#pragma unroll
  for (int mi = 0; mi < 2; ++mi)
#pragma unroll
    for (int ni = 0; ni < 2; ++ni)
#pragma unroll
      for (int g = 0; g < 8; ++g) {
        int gm = m0 + rw * 32 + mi * 16 + g + ((lane & 16) ? 8 : 0);
        int gn = n0 + cw * 32 + ni * 16 + (lane & 15);
        float v = (acc[mi][ni][g] + bias[gn]) * alpha;
        if (mode == 2) {
          ((float*)dst)[(size_t)gm * N + gn] = v;
        } else {
          int bb = gm / Ss, sI = gm % Ss;
          int hh = gn / DHh, dh = gn % DHh;
          size_t idx =
              (mode == 0)
                  ? (((size_t)(bb * Hh + hh) * Ss + sI) * DHh + dh)   // [B,H,S,DH]
                  : (((size_t)(bb * Hh + hh) * DHh + dh) * Ss + sI);  // [B,H,DH,S]
          ((_Float16*)dst)[idx] = (_Float16)v;
        }
      }
}

// ---------------------------------------------------------------------------
// Flash attention: block = 4 waves = 64 query rows of one (b,h).
// K/V chunks (32 keys) are staged global->LDS with async b128 copies,
// double-buffered; all 4 waves share the tiles. Q is pre-scaled by 1/8 and
// held resident in VGPRs. Online softmax; P re-layout C->A via per-wave LDS.
// ---------------------------------------------------------------------------
__global__ __launch_bounds__(128)
void attn_kernel(const _Float16* __restrict__ Qh, const _Float16* __restrict__ Kh,
                 const _Float16* __restrict__ VTh, const unsigned* __restrict__ maskb,
                 _Float16* __restrict__ attO) {
  // Padded row stride 72 halfs (144B): 16B-aligned fragment loads and the 16
  // row addresses of a fragment read map to 16 distinct LDS bank groups.
  __shared__ _Float16 Klds[2][32 * 72];   // [key][d]
  __shared__ _Float16 Vlds[2][64 * 72];   // [d][key]  (from VT layout)
  __shared__ _Float16 Plds[4][16 * 40];   // per-wave P staging

  const int t = threadIdx.x;
  const int wave = t >> 5, lane = t & 31;
  const int tilesPerS = Ss / 64;
  const int bh = blockIdx.x / tilesPerS;
  const int q0 = (blockIdx.x % tilesPerS) * 64 + wave * 16;
  const int b = bh / Hh, h = bh % Hh;

  const _Float16* Qb = Qh + (size_t)bh * Ss * DHh;
  const _Float16* Kb = Kh + (size_t)bh * Ss * DHh;
  const _Float16* Vb = VTh + (size_t)bh * DHh * Ss;

  // Async-copy mapping: 256 16B segments per tile, 2 per thread per tile.
  // K tile: 32 rows x 128B (contiguous in global). V tile: 64 rows x 64B.
  auto issue_chunk = [&](int kbase, int buf) {
#pragma unroll
    for (int i = 0; i < 2; ++i) {
      const int s = t + i * 128;
      const int kr = s >> 3, kp = s & 7;    // K: row(key) 0..31, 16B part 0..7
      async_copy_b128((unsigned)(uintptr_t)&Klds[buf][kr * 72 + kp * 8],
                      Kb + (size_t)(kbase + kr) * DHh + kp * 8);
      const int vr = s >> 2, vp = s & 3;    // V: row(d) 0..63, 16B part 0..3
      async_copy_b128((unsigned)(uintptr_t)&Vlds[buf][vr * 72 + vp * 8],
                      Vb + (size_t)vr * Ss + kbase + vp * 8);
    }
  };

  // resident Q fragments (two 32-wide K chunks over DH=64)
  const int qrow = q0 + (lane & 15);
  v16h aQ[2];
#pragma unroll
  for (int dc = 0; dc < 2; ++dc)
    aQ[dc] = load_a_frag(Qb + (size_t)qrow * DHh + dc * 32, lane);

  float mrow[8], lrow[8];
  v8f acc[4];
  int rowg[8];
  const unsigned* mbp[8];
#pragma unroll
  for (int g = 0; g < 8; ++g) {
    mrow[g] = -3.0e38f;
    lrow[g] = 0.0f;
    rowg[g] = q0 + g + ((lane & 16) ? 8 : 0);
    mbp[g] = maskb + (size_t)(b * Ss + rowg[g]) * WPR;
  }
#pragma unroll
  for (int ni = 0; ni < 4; ++ni)
#pragma unroll
    for (int g = 0; g < 8; ++g) acc[ni][g] = 0.0f;

  issue_chunk(0, 0);   // prologue: chunk 0 -> buffer 0 (4 async ops/lane)

  for (int kc = 0; kc < Ss / 32; ++kc) {
    const int kbase = kc * 32;
    const int buf = kc & 1;

    if (kc + 1 < Ss / 32) {
      issue_chunk(kbase + 32, buf ^ 1);
      // 8 outstanding -> wait to 4: chunk kc's copies are complete (in-order).
      asm volatile("s_wait_asynccnt 0x4" ::: "memory");
    } else {
      asm volatile("s_wait_asynccnt 0x0" ::: "memory");
    }
    __syncthreads();   // chunk kc visible to all 4 waves

    // ---- scores: two 16x16 tiles over this 32-key chunk (K from LDS) ----
    v8f s[2];
#pragma unroll
    for (int tt = 0; tt < 2; ++tt) {
#pragma unroll
      for (int g = 0; g < 8; ++g) s[tt][g] = 0.0f;
      const int krow = tt * 16 + (lane & 15);
#pragma unroll
      for (int dc = 0; dc < 2; ++dc) {
        v16h bK = load_b_frag(
            &Klds[buf][krow * 72 + dc * 32 + ((lane & 16) ? 16 : 0)]);
        s[tt] = wmma_f16(aQ[dc], bK, s[tt]);
      }
    }

    // ---- mask (bit-packed; bit = tt*16 + lane%16 within this word) ----
    unsigned mw[8];
#pragma unroll
    for (int g = 0; g < 8; ++g) mw[g] = mbp[g][kc];
#pragma unroll
    for (int tt = 0; tt < 2; ++tt)
#pragma unroll
      for (int g = 0; g < 8; ++g)
        if (!((mw[g] >> (tt * 16 + (lane & 15))) & 1u)) s[tt][g] = -1.0e9f;

    // ---- online softmax (row reductions over 16-lane halves) ----
#pragma unroll
    for (int g = 0; g < 8; ++g) {
      float v = fmaxf(s[0][g], s[1][g]);
      v = fmaxf(v, __shfl_xor(v, 1));
      v = fmaxf(v, __shfl_xor(v, 2));
      v = fmaxf(v, __shfl_xor(v, 4));
      v = fmaxf(v, __shfl_xor(v, 8));
      float mn = fmaxf(mrow[g], v);
      float r = __expf(mrow[g] - mn);   // 0 on first chunk (mrow = -3e38)
      mrow[g] = mn;
      lrow[g] *= r;
#pragma unroll
      for (int ni = 0; ni < 4; ++ni) acc[ni][g] *= r;

      float p0 = __expf(s[0][g] - mn);
      float p1 = __expf(s[1][g] - mn);
      float ps = p0 + p1;
      ps += __shfl_xor(ps, 1);
      ps += __shfl_xor(ps, 2);
      ps += __shfl_xor(ps, 4);
      ps += __shfl_xor(ps, 8);
      lrow[g] += ps;

      const int prow = g + ((lane & 16) ? 8 : 0);
      Plds[wave][prow * 40 + (lane & 15)]      = (_Float16)p0;
      Plds[wave][prow * 40 + 16 + (lane & 15)] = (_Float16)p1;
    }

    // ---- P @ V over this chunk (V from LDS) ----
    v16h aP = load_a_frag(&Plds[wave][(lane & 15) * 40], lane);
#pragma unroll
    for (int ni = 0; ni < 4; ++ni) {
      v16h bV = load_b_frag(
          &Vlds[buf][(ni * 16 + (lane & 15)) * 72 + ((lane & 16) ? 16 : 0)]);
      acc[ni] = wmma_f16(aP, bV, acc[ni]);
    }

    __syncthreads();   // all waves done reading buf before it is re-filled
  }

  // ---- normalize and store merged-head f16 [B*S, D] ----
  float inv[8];
#pragma unroll
  for (int g = 0; g < 8; ++g) inv[g] = 1.0f / lrow[g];
#pragma unroll
  for (int ni = 0; ni < 4; ++ni)
#pragma unroll
    for (int g = 0; g < 8; ++g) {
      const int dcol = ni * 16 + (lane & 15);
      attO[((size_t)b * Ss + rowg[g]) * Dd + h * DHh + dcol] =
          (_Float16)(acc[ni][g] * inv[g]);
    }
}

// ---------------------------------------------------------------------------
// Launch. Workspace layout (bytes), total ~74 MB:
//   [0)            4 * D*D f16      converted weights (Wq|Wk|Wv|Wo)
//   [+8MiB)        B*H*S*DH f16     Q    (pre-scaled by 1/8)
//   [+16MiB each)  K, VT, attO
//   [tail)         B*S*(S/32) u32   packed mask
// ---------------------------------------------------------------------------
extern "C" void kernel_launch(void* const* d_in, const int* in_sizes, int n_in,
                              void* d_out, int out_size, void* d_ws, size_t ws_size,
                              hipStream_t stream) {
  const float* x   = (const float*)d_in[0];
  const float* y   = (const float*)d_in[1];
  const int*   msk = (const int*)d_in[2];
  const float* Wq  = (const float*)d_in[3];
  const float* bq  = (const float*)d_in[4];
  const float* Wk  = (const float*)d_in[5];
  const float* bk  = (const float*)d_in[6];
  const float* Wv  = (const float*)d_in[7];
  const float* bv  = (const float*)d_in[8];
  const float* Wo  = (const float*)d_in[9];
  const float* bo  = (const float*)d_in[10];

  char* ws = (char*)d_ws;
  const size_t WSZ  = (size_t)Dd * Dd * sizeof(_Float16);            // 2 MiB
  const size_t QSZ  = (size_t)Bb * Hh * Ss * DHh * sizeof(_Float16); // 16 MiB
  _Float16* Wh   = (_Float16*)(ws);                 // Wq|Wk|Wv|Wo packed
  _Float16* Wqh  = Wh;
  _Float16* Wkh  = Wh + (size_t)Dd * Dd;
  _Float16* Wvh  = Wh + (size_t)2 * Dd * Dd;
  _Float16* Woh  = Wh + (size_t)3 * Dd * Dd;
  _Float16* Qh   = (_Float16*)(ws + 4 * WSZ);
  _Float16* Kh   = (_Float16*)(ws + 4 * WSZ + QSZ);
  _Float16* VTh  = (_Float16*)(ws + 4 * WSZ + 2 * QSZ);
  _Float16* attO = (_Float16*)(ws + 4 * WSZ + 3 * QSZ);
  unsigned* maskb = (unsigned*)(ws + 4 * WSZ + 4 * QSZ);

  const int M = Bb * Ss;                         // 8192
  const int gemmBlocks = (M / 64) * (Dd / 64);   // 2048

  cvt_w_kernel<<<(4 * Dd * Dd) / 256, 256, 0, stream>>>(Wq, Wk, Wv, Wo, Wh);
  pack_mask_kernel<<<(Bb * Ss * WPR) / 256, 256, 0, stream>>>(msk, maskb);

  // Q = (x@Wq + bq) * 1/sqrt(64);  K = y@Wk + bk;  VT = (y@Wv + bv)^T per head
  gemm64_kernel<0><<<gemmBlocks, 128, 0, stream>>>(x, Wqh, bq, Qh, M, Dd, Dd, 0, 0.125f);
  gemm64_kernel<0><<<gemmBlocks, 128, 0, stream>>>(y, Wkh, bk, Kh, M, Dd, Dd, 0, 1.0f);
  gemm64_kernel<0><<<gemmBlocks, 128, 0, stream>>>(y, Wvh, bv, VTh, M, Dd, Dd, 1, 1.0f);

  attn_kernel<<<Bb * Hh * (Ss / 64), 128, 0, stream>>>(Qh, Kh, VTh, maskb, attO);

  // out = attO @ Wo + bo  (fp32)
  gemm64_kernel<1><<<gemmBlocks, 128, 0, stream>>>(attO, Woh, bo, d_out, M, Dd, Dd, 2, 1.0f);
}
